// CRF_10299331575835
// MI455X (gfx1250) — matrix operand order repaired and checked by reading, not compiled, not run on hardware
//
#include <hip/hip_runtime.h>
#include <hip/hip_bf16.h>

typedef __attribute__((ext_vector_type(16))) _Float16 v16h;
typedef __attribute__((ext_vector_type(8)))  _Float16 v8h;
typedef __attribute__((ext_vector_type(8)))  float    v8f;

#define CRF_T 512
#define CRF_K 32
#define START_IDX 0
#define END_IDX 1
#define LN2F 0.6931471805599453f

// ---------------------------------------------------------------------------
// DPP16 butterfly reductions within each 16-lane row of the wave32.
// dpp_ctrl: 0xB1 = quad_perm xor1, 0x4E = quad_perm xor2,
//           0x141 = row_half_mirror (xor7), 0x140 = row_mirror (xor15).
// {1,2,7,15} generate the full 16-element xor group, so max/sum butterflies
// converge in 4 pure-VALU steps (no LDS, no waits).
// ---------------------------------------------------------------------------
__device__ __forceinline__ float rowmax16(float v) {
    int i;
    i = __builtin_amdgcn_update_dpp(0, __float_as_int(v), 0xB1,  0xF, 0xF, true);
    v = fmaxf(v, __int_as_float(i));
    i = __builtin_amdgcn_update_dpp(0, __float_as_int(v), 0x4E,  0xF, 0xF, true);
    v = fmaxf(v, __int_as_float(i));
    i = __builtin_amdgcn_update_dpp(0, __float_as_int(v), 0x141, 0xF, 0xF, true);
    v = fmaxf(v, __int_as_float(i));
    i = __builtin_amdgcn_update_dpp(0, __float_as_int(v), 0x140, 0xF, 0xF, true);
    v = fmaxf(v, __int_as_float(i));
    return v;
}

__device__ __forceinline__ float rowsum16(float v) {
    int i;
    i = __builtin_amdgcn_update_dpp(0, __float_as_int(v), 0xB1,  0xF, 0xF, true);
    v += __int_as_float(i);
    i = __builtin_amdgcn_update_dpp(0, __float_as_int(v), 0x4E,  0xF, 0xF, true);
    v += __int_as_float(i);
    i = __builtin_amdgcn_update_dpp(0, __float_as_int(v), 0x141, 0xF, 0xF, true);
    v += __int_as_float(i);
    i = __builtin_amdgcn_update_dpp(0, __float_as_int(v), 0x140, 0xF, 0xF, true);
    v += __int_as_float(i);
    return v;
}

// ---------------------------------------------------------------------------
// Kernel 1: forward algorithm (log-partition) via scaled linear recursion.
// One wave (32 lanes) handles 16 batch rows. Per time step:
//   s(16x32) = a(16x32,f16) x E(32x32,f16)  -> two V_WMMA_F32_16X16X32_F16
//   p = s * exp(logit_t); exact power-of-two row rescale via frexp/ldexp;
//   rows past their sequence length are frozen (branchless selects).
// E = exp(transitions) lives in registers (two v16h B operands) for all T.
// The running scale is an integer exponent per row; only the finalization
// needs a v_log_f32.
// ---------------------------------------------------------------------------
__global__ __launch_bounds__(32) void crf_forward_kernel(
    const float* __restrict__ logits,      // (B, T, 32)
    const int*   __restrict__ lens,        // (B,)
    const float* __restrict__ trans,       // (32, 32)
    float*       __restrict__ out_norm)    // (B,) log-partition
{
    __shared__ __attribute__((aligned(32))) _Float16 stage[16 * CRF_K]; // 16x32 f16 tile

    const int lane = threadIdx.x;          // 0..31 (wave32)
    const int hi   = lane >> 4;            // lane-half: 0 or 1
    const int lj   = lane & 15;            // lane within half
    const int base = blockIdx.x * 16;      // first batch row of this wave

    // --- Build B operands: E[k][j] = exp(trans[k][j]) in WMMA B layout.
    // B (32x16 f16), contiguous k order chosen here: lanes 0-15 hold K=0..15, lanes 16-31 hold K=16..31,
    // column N = lane%16, half e <-> K = e + 16*hi.
    v16h Bt0, Bt1;
#pragma unroll
    for (int e = 0; e < 16; ++e) {
        const int k = e + 16 * hi;
        Bt0[e] = (_Float16)__expf(trans[k * CRF_K + lj]);
        Bt1[e] = (_Float16)__expf(trans[k * CRF_K + lj + 16]);
    }
    // END-transition weights, indexed by column j of the D layout.
    const float eend0 = __expf(trans[(lj)      * CRF_K + END_IDX]);
    const float eend1 = __expf(trans[(lj + 16) * CRF_K + END_IDX]);

    // Sequence lengths per D-layout row slot r (row m = r + 8*hi).
    int lenv[8];
#pragma unroll
    for (int r = 0; r < 8; ++r) lenv[r] = lens[base + r + 8 * hi];

    // State in D layout: a0 = cols 0..15, a1 = cols 16..31.
    // cexp = accumulated power-of-two scale exponent (alpha = log a + cexp*ln2).
    // alpha0 = onehot(START) in linear space.
    float a0[8], a1[8];
    int   cexp[8];
#pragma unroll
    for (int r = 0; r < 8; ++r) {
        a0[r]   = (lj == START_IDX) ? 1.0f : 0.0f;
        a1[r]   = 0.0f;
        cexp[r] = 0;
    }

    const float* lg = logits + (size_t)base * CRF_T * CRF_K;

    for (int t = 0; t < CRF_T; ++t) {
        // 1) Stage current alpha (linear, normalized) as f16 row-major 16x32.
#pragma unroll
        for (int r = 0; r < 8; ++r) {
            const int m = r + 8 * hi;
            stage[m * CRF_K + lj]      = (_Float16)a0[r];
            stage[m * CRF_K + lj + 16] = (_Float16)a1[r];
        }
        asm volatile("s_wait_dscnt 0" ::: "memory");

        // 2) Load A operand: lane = row m = lane%16; halves 0..15 <-> K = 16*hi + e
        //    (contiguous, matching B). Two b128 loads.
        const v8h alo = *(const v8h*)&stage[lj * CRF_K + hi * 16];
        const v8h ahi = *(const v8h*)&stage[lj * CRF_K + hi * 16 + 8];
        union { v16h v; v8h h[2]; } ua;
        ua.h[0] = alo; ua.h[1] = ahi;
        const v16h A = ua.v;

        // 3) s = a x E  (fresh accumulation from zero each step)
        const v8f z = {0.f, 0.f, 0.f, 0.f, 0.f, 0.f, 0.f, 0.f};
        v8f s0 = __builtin_amdgcn_wmma_f32_16x16x32_f16(
            false, A, false, Bt0, (short)0, z, false, false);
        v8f s1 = __builtin_amdgcn_wmma_f32_16x16x32_f16(
            false, A, false, Bt1, (short)0, z, false, false);

        // 4) Fold emissions: p = s * exp(logit[t]).  D layout addressing.
        float p0[8], p1[8];
#pragma unroll
        for (int r = 0; r < 8; ++r) {
            const int m = r + 8 * hi;
            const float* row = lg + ((size_t)m * CRF_T + t) * CRF_K;
            if (r == 0 && t + 1 < CRF_T)
                __builtin_prefetch(row + CRF_K + lj, 0, 1);  // next-t line
            p0[r] = s0[r] * __expf(row[lj]);
            p1[r] = s1[r] * __expf(row[lj + 16]);
        }

        // 5) Row-max over 32 cols (both tiles + 16 lanes of the half), then
        //    exact power-of-two rescale (no log/rcp), branchless freeze.
#pragma unroll
        for (int r = 0; r < 8; ++r) {
            const float rm = rowmax16(fmaxf(p0[r], p1[r]));
            const int   e  = __builtin_amdgcn_frexp_expf(rm); // rm = m*2^e, m in [0.5,1)
            const bool alive = (lenv[r] > t);
            const float na0 = __builtin_amdgcn_ldexpf(p0[r], -e);
            const float na1 = __builtin_amdgcn_ldexpf(p1[r], -e);
            a0[r]   = alive ? na0 : a0[r];
            a1[r]   = alive ? na1 : a1[r];
            cexp[r] = alive ? (cexp[r] + e) : cexp[r];
        }
    }

    // Final: norm = ln2 * ( log2(sum_j a[j]*exp(trans[j][END])) + cexp ), per row.
#pragma unroll
    for (int r = 0; r < 8; ++r) {
        const float v = rowsum16(a0[r] * eend0 + a1[r] * eend1);
        if (lj == 0)
            out_norm[base + r + 8 * hi] =
                LN2F * (__builtin_amdgcn_logf(v) + (float)cexp[r]);
    }
}

// ---------------------------------------------------------------------------
// Kernel 2: gold-path score; one wave per batch. Subtracts from the norm
// already in d_out (stream-ordered after kernel 1).
// ---------------------------------------------------------------------------
__global__ __launch_bounds__(256) void crf_path_kernel(
    const float* __restrict__ logits,   // (B, T, 32)
    const int*   __restrict__ y,        // (B, T)
    const int*   __restrict__ lens,     // (B,)
    const float* __restrict__ trans,    // (32, 32)
    float*       __restrict__ out)      // (B,) norm -> norm - score
{
    __shared__ float tr[CRF_K * CRF_K];
    for (int i = threadIdx.x; i < CRF_K * CRF_K; i += 256) tr[i] = trans[i];
    __syncthreads();

    const int wave = threadIdx.x >> 5;
    const int lane = threadIdx.x & 31;
    const int b    = blockIdx.x * 8 + wave;

    const int len = lens[b];
    const int*   yb = y      + (size_t)b * CRF_T;
    const float* lb = logits + (size_t)b * CRF_T * CRF_K;

    float acc = 0.0f;
    for (int i = lane; i <= CRF_T; i += 32) {
        // transition term for extended-label position i -> i+1
        if (i < len + 1) {
            const int la  = (i == 0) ? START_IDX : yb[i - 1];
            const int lb2 = (i < len) ? yb[i] : END_IDX;   // i+1 >= len+1 -> END
            acc += tr[la * CRF_K + lb2];
        }
        // emission term for t = i
        if (i < CRF_T && i < len) {
            acc += lb[(size_t)i * CRF_K + yb[i]];
        }
    }
    // wave32 sum: DPP within 16-lane rows, then combine the two halves.
    acc = rowsum16(acc);
    const float h0 = __int_as_float(__builtin_amdgcn_readlane(__float_as_int(acc), 0));
    const float h1 = __int_as_float(__builtin_amdgcn_readlane(__float_as_int(acc), 16));
    if (lane == 0) out[b] = out[b] - (h0 + h1);
}

// ---------------------------------------------------------------------------
extern "C" void kernel_launch(void* const* d_in, const int* in_sizes, int n_in,
                              void* d_out, int out_size, void* d_ws, size_t ws_size,
                              hipStream_t stream) {
    const float* logits = (const float*)d_in[0];   // (B, T, 32) f32
    const int*   y_ent  = (const int*)  d_in[1];   // (B, T) i32
    const int*   lens   = (const int*)  d_in[2];   // (B,) i32
    const float* trans  = (const float*)d_in[3];   // (32, 32) f32
    float* out = (float*)d_out;                    // (B,) f32

    const int B = in_sizes[2];                     // 2048

    crf_forward_kernel<<<B / 16, 32, 0, stream>>>(logits, lens, trans, out);
    crf_path_kernel<<<B / 8, 256, 0, stream>>>(logits, y_ent, lens, trans, out);
}
